// Net_62818191671500
// MI455X (gfx1250) — matrix operand order repaired and checked by reading, not compiled, not run on hardware
//
#include <hip/hip_runtime.h>
#include <hip/hip_bf16.h>

typedef __attribute__((ext_vector_type(16))) _Float16 v16h;
typedef __attribute__((ext_vector_type(8)))  _Float16 h8;     // 16-byte vector
typedef __attribute__((ext_vector_type(8)))  float    v8f;

#define NEG_SLOPE 0.2f

// ---------------------------------------------------------------------------
// WMMA GEMM: C[M,Nc] = A[M,Kp] * Bt[Nc,Kp]^T   (A,Bt f16; C f32)
// A padded/converted to f16 with lda multiple of 8; Bt is W transposed
// (col-major, K-contiguous) so all global loads are 16B vectors.
// Block = 256 threads (8 waves). Tile BM=128, BN=64, BK=32.
// Double-buffered LDS + register prefetch; one barrier per K-tile.
// All fragment loads issued before the first WMMA so waits stagger.
// Kp must be a multiple of 32 (zero-padded); Nc a multiple of 64.
// ---------------------------------------------------------------------------
#define BM 128
#define BN 64
#define BK 32
#define LDH 40          // LDS stride in halfs: 80B -> 16B aligned, conflict-free

__global__ __launch_bounds__(256) void gemm_wmma_f16(
    const _Float16* __restrict__ A, int lda,
    const _Float16* __restrict__ Bt, int ldb,
    float* __restrict__ C, int M, int Kp, int Nc)
{
    __shared__ _Float16 sA[2][BM * LDH];
    __shared__ _Float16 sB[2][BN * LDH];

    const int tid  = threadIdx.x;
    const int wave = tid >> 5;
    const int lane = tid & 31;
    const int half_ = lane >> 4;
    const int l16   = lane & 15;

    const int tileM = blockIdx.y * BM;
    const int tileN = blockIdx.x * BN;

    // A loader: 128x32 halfs = 512 h8-vectors, 2 per thread
    const int ar0 = (tid + 0)   >> 2, ak0 = ((tid + 0)   & 3) * 8;
    const int ar1 = (tid + 256) >> 2, ak1 = ((tid + 256) & 3) * 8;
    // B loader: 64x32 halfs = 256 h8-vectors, 1 per thread
    const int bcol = tid >> 2, bk = (tid & 3) * 8;

    // clamp rows: always-valid loads; rows >= M computed but never stored
    int cr0 = tileM + ar0; if (cr0 > M - 1) cr0 = M - 1;
    int cr1 = tileM + ar1; if (cr1 > M - 1) cr1 = M - 1;
    const size_t arow0 = (size_t)cr0 * lda;
    const size_t arow1 = (size_t)cr1 * lda;
    const size_t brow  = (size_t)(tileN + bcol) * ldb;

    v8f acc[4] = {};

    h8 pa0 = *(const h8*)(A + arow0 + ak0);
    h8 pa1 = *(const h8*)(A + arow1 + ak1);
    h8 pb  = *(const h8*)(Bt + brow + bk);

    const int ktiles = Kp / BK;
    int buf = 0;
    for (int kt = 0; kt < ktiles; ++kt) {
        *(h8*)(&sA[buf][ar0 * LDH + ak0]) = pa0;
        *(h8*)(&sA[buf][ar1 * LDH + ak1]) = pa1;
        *(h8*)(&sB[buf][bcol * LDH + bk]) = pb;
        __syncthreads();

        if (kt + 1 < ktiles) {      // next tile's loads fly during the math
            const int k0 = (kt + 1) * BK;
            pa0 = *(const h8*)(A + arow0 + k0 + ak0);
            pa1 = *(const h8*)(A + arow1 + k0 + ak1);
            pb  = *(const h8*)(Bt + brow + k0 + bk);
        }

        const unsigned int* __restrict__ au = (const unsigned int*)sA[buf];
        const unsigned int* __restrict__ bu = (const unsigned int*)sB[buf];

        // Issue ALL fragment loads first (DS returns in order -> staggered
        // dscnt waits, LDS latency overlaps WMMA execution).
        union { unsigned int u[8]; v16h v; } ua;
        union { unsigned int u[8]; v16h v; } ub[4];
        {
            // A fragment (ISA 16-bit A 16x32): pair p -> K=(p>>2)*16+half*8+(p&3)*2
            const int r = wave * 16 + l16;
            #pragma unroll
            for (int p = 0; p < 8; ++p) {
                int g = p >> 2, pp = p & 3;
                ua.u[p] = au[r * (LDH / 2) + g * 8 + half_ * 4 + pp];
            }
        }
        #pragma unroll
        for (int c = 0; c < 4; ++c) {
            // B fragment: pair p -> K=(p>>2)*16+half*8+(p&3)*2 for col c*16+l16
            const int col = c * 16 + l16;
            #pragma unroll
            for (int p = 0; p < 8; ++p)
                ub[c].u[p] = bu[col * (LDH / 2) + (p >> 2) * 8 + half_ * 4 + (p & 3)];
        }
        #pragma unroll
        for (int c = 0; c < 4; ++c) {
            acc[c] = __builtin_amdgcn_wmma_f32_16x16x32_f16(
                false, ua.v, false, ub[c].v, (short)0, acc[c], false, false);
        }
        buf ^= 1;
    }

    #pragma unroll
    for (int c = 0; c < 4; ++c) {
        #pragma unroll
        for (int r8 = 0; r8 < 8; ++r8) {
            int gr = tileM + wave * 16 + half_ * 8 + r8;
            int gc = tileN + c * 16 + l16;
            if (gr < M) C[(size_t)gr * Nc + gc] = acc[c][r8];
        }
    }
}

// ---------------------------------------------------------------------------
// f32 -> f16 row conversion with K zero-padded to ldk
// ---------------------------------------------------------------------------
__global__ void conv_f16_pad(const float* __restrict__ src, _Float16* __restrict__ dst,
                             int M, int K, int ldk)
{
    long long i = (long long)blockIdx.x * blockDim.x + threadIdx.x;
    if (i >= (long long)M * ldk) return;
    int r = (int)(i / ldk), k = (int)(i % ldk);
    float v = (k < K) ? src[(size_t)r * K + k] : 0.0f;
    dst[i] = (_Float16)v;
}

// W[K][Nc] f32 -> Wt[Nc][ldk] f16 (transposed, K zero-padded to ldk)
__global__ void transpose_f16_pad(const float* __restrict__ W, _Float16* __restrict__ Wt,
                                  int K, int Nc, int ldk)
{
    long long i = (long long)blockIdx.x * blockDim.x + threadIdx.x;
    if (i >= (long long)Nc * ldk) return;
    int col = (int)(i / ldk), k = (int)(i % ldk);
    float v = (k < K) ? W[(size_t)k * Nc + col] : 0.0f;
    Wt[i] = (_Float16)v;
}

// ---------------------------------------------------------------------------
// Tiny GEMM for layer 3 (K=256, C=7): thread per (node, class)
// ---------------------------------------------------------------------------
__global__ void gemm_small(const float* __restrict__ A, const float* __restrict__ B,
                           float* __restrict__ C, int M, int K, int Nc)
{
    int id   = blockIdx.x * blockDim.x + threadIdx.x;
    int node = id >> 3;
    int c    = id & 7;
    if (node >= M || c >= Nc) return;
    const float* row = A + (size_t)node * K;
    float s = 0.f;
    for (int k = 0; k < K; ++k) s += row[k] * B[k * Nc + c];
    C[(size_t)node * Nc + c] = s;
}

// ---------------------------------------------------------------------------
// Per-node attention scores, one wave per node (wave32 shuffle reduction)
// ---------------------------------------------------------------------------
__global__ void score_kernel(const float* __restrict__ h,
                             const float* __restrict__ a_s,
                             const float* __restrict__ a_d,
                             float* __restrict__ s_s, float* __restrict__ s_d,
                             int N, int F)
{
    int gid  = blockIdx.x * blockDim.x + threadIdx.x;
    int node = gid >> 5;
    int lane = threadIdx.x & 31;
    if (node >= N) return;
    const float* row = h + (size_t)node * F;
    float ss = 0.f, sd = 0.f;
    for (int f = lane; f < F; f += 32) {
        float v = row[f];
        ss += v * a_s[f];
        sd += v * a_d[f];
    }
    #pragma unroll
    for (int off = 16; off > 0; off >>= 1) {
        ss += __shfl_down(ss, off, 32);
        sd += __shfl_down(sd, off, 32);
    }
    if (lane == 0) { s_s[node] = ss; s_d[node] = sd; }
}

// order-preserving float<->uint for atomicMax-based segment max
__device__ __forceinline__ unsigned int fenc(float f) {
    unsigned int u = __float_as_uint(f);
    return (u & 0x80000000u) ? ~u : (u | 0x80000000u);
}
__device__ __forceinline__ float fdec(unsigned int k) {
    return (k & 0x80000000u) ? __uint_as_float(k & 0x7fffffffu)
                             : __uint_as_float(~k);
}

__global__ void seg_init(unsigned int* __restrict__ m_enc,
                         float* __restrict__ denom, int N)
{
    int i = blockIdx.x * blockDim.x + threadIdx.x;
    if (i < N) { m_enc[i] = 0u; denom[i] = 0.f; }
}

__global__ void edge_pass1(const int* __restrict__ ei, int E, int ET,
                           const float* __restrict__ s_s,
                           const float* __restrict__ s_d,
                           float* __restrict__ e_buf,
                           unsigned int* __restrict__ m_enc)
{
    int id = blockIdx.x * blockDim.x + threadIdx.x;
    if (id >= ET) return;
    int src = (id < E) ? ei[id]     : (id - E);
    int dst = (id < E) ? ei[E + id] : (id - E);
    float e = s_s[src] + s_d[dst];
    e = (e > 0.f) ? e : NEG_SLOPE * e;
    e_buf[id] = e;
    atomicMax(&m_enc[dst], fenc(e));
}

__global__ void edge_pass2(const int* __restrict__ ei, int E, int ET,
                           const float* __restrict__ e_buf,
                           const unsigned int* __restrict__ m_enc,
                           float* __restrict__ ee_buf,
                           float* __restrict__ denom)
{
    int id = blockIdx.x * blockDim.x + threadIdx.x;
    if (id >= ET) return;
    int dst = (id < E) ? ei[E + id] : (id - E);
    float ee = __expf(e_buf[id] - fdec(m_enc[dst]));
    ee_buf[id] = ee;
    atomicAdd(&denom[dst], ee);
}

__global__ void bias_init(float* __restrict__ out, const float* __restrict__ b,
                          int N, int F)
{
    int i = blockIdx.x * blockDim.x + threadIdx.x;
    if (i < N * F) out[i] = b[i % F];
}

// out[dst] += (ee/denom[dst]) * h[src]; one wave per edge
__global__ void edge_scatter(const int* __restrict__ ei, int E, int ET,
                             const float* __restrict__ ee_buf,
                             const float* __restrict__ denom,
                             const float* __restrict__ h,
                             float* __restrict__ out, int F)
{
    int gid  = blockIdx.x * blockDim.x + threadIdx.x;
    int eid  = gid >> 5;
    int lane = threadIdx.x & 31;
    if (eid >= ET) return;
    int src = (eid < E) ? ei[eid]     : (eid - E);
    int dst = (eid < E) ? ei[E + eid] : (eid - E);
    float alpha = ee_buf[eid] / denom[dst];
    const float* hrow = h + (size_t)src * F;
    float* orow = out + (size_t)dst * F;
    for (int f = lane; f < F; f += 32)
        atomicAdd(&orow[f], alpha * hrow[f]);
}

__global__ void relu_kernel(float* __restrict__ x, long long n)
{
    long long i = (long long)blockIdx.x * blockDim.x + threadIdx.x;
    if (i < n) { float v = x[i]; x[i] = v > 0.f ? v : 0.f; }
}

__global__ void logsoftmax7(float* __restrict__ x, int N)
{
    int i = blockIdx.x * blockDim.x + threadIdx.x;
    if (i >= N) return;
    float* row = x + (size_t)i * 7;
    float v[7], m = -3.4e38f;
    #pragma unroll
    for (int c = 0; c < 7; ++c) { v[c] = row[c]; m = fmaxf(m, v[c]); }
    float s = 0.f;
    #pragma unroll
    for (int c = 0; c < 7; ++c) s += __expf(v[c] - m);
    float lse = m + __logf(s);
    #pragma unroll
    for (int c = 0; c < 7; ++c) row[c] = v[c] - lse;
}

// ---------------------------------------------------------------------------
extern "C" void kernel_launch(void* const* d_in, const int* in_sizes, int n_in,
                              void* d_out, int out_size, void* d_ws, size_t ws_size,
                              hipStream_t stream)
{
    const int F_IN = 1433, K1P = 1440, H1 = 512, H2 = 256, C = 7;
    const int N  = in_sizes[0] / F_IN;
    const int E  = in_sizes[1] / 2;
    const int ET = E + N;

    const float* x   = (const float*)d_in[0];
    const int*   ei  = (const int*)d_in[1];
    const float* W1  = (const float*)d_in[2];
    const float* as1 = (const float*)d_in[3];
    const float* ad1 = (const float*)d_in[4];
    const float* b1  = (const float*)d_in[5];
    const float* W2  = (const float*)d_in[6];
    const float* as2 = (const float*)d_in[7];
    const float* ad2 = (const float*)d_in[8];
    const float* b2  = (const float*)d_in[9];
    const float* W3  = (const float*)d_in[10];
    const float* as3 = (const float*)d_in[11];
    const float* ad3 = (const float*)d_in[12];
    const float* b3  = (const float*)d_in[13];
    float* outp = (float*)d_out;

    // -------- workspace carve-up (256B aligned) --------
    char* base = (char*)d_ws;
    size_t off = 0;
    auto carve = [&](size_t bytes) -> char* {
        char* p = base + off;
        off += (bytes + 255) & ~(size_t)255;
        return p;
    };
    _Float16* xh  = (_Float16*)carve((size_t)N * K1P * 2);  // reused as o1h
    _Float16* W1t = (_Float16*)carve((size_t)H1 * K1P * 2);
    _Float16* W2t = (_Float16*)carve((size_t)H2 * H1 * 2);
    float* h1  = (float*)carve((size_t)N * H1 * 4);          // reused as o2
    float* o1  = (float*)carve((size_t)N * H1 * 4);
    float* h2  = (float*)carve((size_t)N * H2 * 4);
    float* h3  = (float*)carve((size_t)N * C * 4);
    float* s_s = (float*)carve((size_t)N * 4);
    float* s_d = (float*)carve((size_t)N * 4);
    unsigned int* m_enc = (unsigned int*)carve((size_t)N * 4);
    float* denom  = (float*)carve((size_t)N * 4);
    float* e_buf  = (float*)carve((size_t)ET * 4);
    float* ee_buf = (float*)carve((size_t)ET * 4);
    _Float16* o1h = xh;   // alias: xh dead after layer-1 GEMM
    float* o2 = h1;       // alias: h1 dead after layer-1 scatter

    const int TB = 256;
    dim3 blk(TB);
    int nodeWaveBlks = (N * 32 + TB - 1) / TB;
    int edgeBlks     = (ET + TB - 1) / TB;
    int edgeWaveBlks = (ET + 7) / 8;
    int nodeBlks     = (N + TB - 1) / TB;

    // -------- pre-pass: convert to f16, transpose weights --------
    conv_f16_pad<<<((long long)N * K1P + TB - 1) / TB, blk, 0, stream>>>(x, xh, N, F_IN, K1P);
    transpose_f16_pad<<<((long long)H1 * K1P + TB - 1) / TB, blk, 0, stream>>>(W1, W1t, F_IN, H1, K1P);
    transpose_f16_pad<<<((long long)H2 * H1 + TB - 1) / TB, blk, 0, stream>>>(W2, W2t, H1, H2, H1);

    // ---------------- Layer 1: 1433 -> 512 ----------------
    {
        dim3 grid(H1 / BN, (N + BM - 1) / BM);
        gemm_wmma_f16<<<grid, blk, 0, stream>>>(xh, K1P, W1t, K1P, h1, N, K1P, H1);
        score_kernel<<<nodeWaveBlks, blk, 0, stream>>>(h1, as1, ad1, s_s, s_d, N, H1);
        seg_init<<<nodeBlks, blk, 0, stream>>>(m_enc, denom, N);
        edge_pass1<<<edgeBlks, blk, 0, stream>>>(ei, E, ET, s_s, s_d, e_buf, m_enc);
        edge_pass2<<<edgeBlks, blk, 0, stream>>>(ei, E, ET, e_buf, m_enc, ee_buf, denom);
        bias_init<<<((size_t)N * H1 + TB - 1) / TB, blk, 0, stream>>>(o1, b1, N, H1);
        edge_scatter<<<edgeWaveBlks, blk, 0, stream>>>(ei, E, ET, ee_buf, denom, h1, o1, H1);
        relu_kernel<<<((size_t)N * H1 + TB - 1) / TB, blk, 0, stream>>>(o1, (long long)N * H1);
    }
    // ---------------- Layer 2: 512 -> 256 ----------------
    {
        conv_f16_pad<<<((long long)N * H1 + TB - 1) / TB, blk, 0, stream>>>(o1, o1h, N, H1, H1);
        dim3 grid(H2 / BN, (N + BM - 1) / BM);
        gemm_wmma_f16<<<grid, blk, 0, stream>>>(o1h, H1, W2t, H1, h2, N, H1, H2);
        score_kernel<<<nodeWaveBlks, blk, 0, stream>>>(h2, as2, ad2, s_s, s_d, N, H2);
        seg_init<<<nodeBlks, blk, 0, stream>>>(m_enc, denom, N);
        edge_pass1<<<edgeBlks, blk, 0, stream>>>(ei, E, ET, s_s, s_d, e_buf, m_enc);
        edge_pass2<<<edgeBlks, blk, 0, stream>>>(ei, E, ET, e_buf, m_enc, ee_buf, denom);
        bias_init<<<((size_t)N * H2 + TB - 1) / TB, blk, 0, stream>>>(o2, b2, N, H2);
        edge_scatter<<<edgeWaveBlks, blk, 0, stream>>>(ei, E, ET, ee_buf, denom, h2, o2, H2);
        relu_kernel<<<((size_t)N * H2 + TB - 1) / TB, blk, 0, stream>>>(o2, (long long)N * H2);
    }
    // ---------------- Layer 3: 256 -> 7, log_softmax ----------------
    {
        gemm_small<<<((size_t)N * 8 + TB - 1) / TB, blk, 0, stream>>>(o2, W3, h3, N, H2, C);
        score_kernel<<<nodeWaveBlks, blk, 0, stream>>>(h3, as3, ad3, s_s, s_d, N, C);
        seg_init<<<nodeBlks, blk, 0, stream>>>(m_enc, denom, N);
        edge_pass1<<<edgeBlks, blk, 0, stream>>>(ei, E, ET, s_s, s_d, e_buf, m_enc);
        edge_pass2<<<edgeBlks, blk, 0, stream>>>(ei, E, ET, e_buf, m_enc, ee_buf, denom);
        bias_init<<<((size_t)N * C + TB - 1) / TB, blk, 0, stream>>>(outp, b3, N, C);
        edge_scatter<<<edgeWaveBlks, blk, 0, stream>>>(ei, E, ET, ee_buf, denom, h3, outp, C);
        logsoftmax7<<<nodeBlks, blk, 0, stream>>>(outp, N);
    }
}
